// MixtralMoeBlock_24885040513469
// MI455X (gfx1250) — compile-verified, matrix operand order checked by reading, not run on hardware
//
#include <hip/hip_runtime.h>
#include <stdint.h>

#define E_EXPERTS 8
#define M_TOK     4096      // B*S
#define H_DIM     1024
#define F_DIM     2048
#define THREADS   256
#define WAVES     8

// per-expert sizes in dwords
#define PWE_DW  ((size_t)H_DIM * F_DIM / 2)   // packed weights (bf16)
#define XG_E_DW ((size_t)M_TOK * H_DIM / 2)   // gathered/packed x rows
#define HP_E_DW ((size_t)M_TOK * F_DIM / 2)   // packed h rows

typedef __attribute__((ext_vector_type(16))) __bf16 v16bf;
typedef __attribute__((ext_vector_type(8)))  float  v8f;

union Frag { v16bf v; uint32_t u[8]; };

__device__ __forceinline__ uint32_t pack_bf16(float a, float b) {
  union { float f; uint32_t u; } ua{a}, ub{b};
  uint32_t lo = (ua.u + 0x7FFFu + ((ua.u >> 16) & 1u)) >> 16;   // RNE
  uint32_t hi = (ub.u + 0x7FFFu + ((ub.u >> 16) & 1u)) >> 16;
  return (lo & 0xFFFFu) | (hi << 16);
}
__device__ __forceinline__ uint16_t f2bf(float a) {
  union { float f; uint32_t u; } ua{a};
  return (uint16_t)((ua.u + 0x7FFFu + ((ua.u >> 16) & 1u)) >> 16);
}
__device__ __forceinline__ void load8(Frag& f, const uint32_t* p) {
  *(uint4*)&f.u[0] = *(const uint4*)(p);
  *(uint4*)&f.u[4] = *(const uint4*)(p + 4);
}

// ---------------------------------------------------------------------------
// Kernel 0: repack weight tensor [E][K][N] f32 -> bf16, pre-swizzled into
// WMMA B-fragment layout:  dword index = ((nt*(K/32) + kt)*32 + lane)*8 + v
// ---------------------------------------------------------------------------
__global__ void repack_kernel(const float* __restrict__ W,
                              uint32_t* __restrict__ P, int K, int N) {
  const int e  = blockIdx.z;
  const int k0 = blockIdx.x * 32;
  const int n0 = blockIdx.y * 128;
  const int tid = threadIdx.x;

  const float* Ws = W + (size_t)e * K * N;
  uint32_t*    Pd = P + (size_t)e * ((size_t)K * N / 2);

  __shared__ float ls[32][129];
  for (int i = tid; i < 32 * 128; i += THREADS) {
    int r = i >> 7, c = i & 127;
    ls[r][c] = Ws[(size_t)(k0 + r) * N + n0 + c];
  }
  __syncthreads();

  const int lane = tid & 31;
  const int ntl  = tid >> 5;
  const int nt   = (n0 >> 4) + ntl;
  uint32_t o[8];
#pragma unroll
  for (int v = 0; v < 8; ++v) {
    int kb = ((v >> 2) << 4) + ((lane >> 4) << 3) + ((v & 3) << 1);
    int n  = (ntl << 4) + (lane & 15);
    o[v] = pack_bf16(ls[kb][n], ls[kb + 1][n]);
  }
  size_t base = ((size_t)nt * (K / 32) + blockIdx.x) * 256 + lane * 8;
  *(uint4*)&Pd[base]     = make_uint4(o[0], o[1], o[2], o[3]);
  *(uint4*)&Pd[base + 4] = make_uint4(o[4], o[5], o[6], o[7]);
}

// ---------------------------------------------------------------------------
// Kernel 1: router (one wave32 per token).
// ---------------------------------------------------------------------------
__global__ void router_kernel(const float* __restrict__ x,
                              const float* __restrict__ gate,
                              int* __restrict__ count,
                              int* __restrict__ rowTsi,
                              float* __restrict__ rowWt) {
  const int wave  = threadIdx.x >> 5;
  const int lane  = threadIdx.x & 31;
  const int token = blockIdx.x * WAVES + wave;
  if (token >= M_TOK) return;

  float p[E_EXPERTS];
#pragma unroll
  for (int e = 0; e < E_EXPERTS; ++e) p[e] = 0.f;

  const float* xr = x + (size_t)token * H_DIM;
  for (int h = lane; h < H_DIM; h += 32) {
    float xv = xr[h];
    const float* g = gate + (size_t)h * E_EXPERTS;
#pragma unroll
    for (int e = 0; e < E_EXPERTS; ++e) p[e] += xv * g[e];
  }
#pragma unroll
  for (int e = 0; e < E_EXPERTS; ++e) {
    float v = p[e];
#pragma unroll
    for (int s = 16; s >= 1; s >>= 1) v += __shfl_xor(v, s, 32);
    p[e] = v;
  }
  if (lane == 0) {
    float b1v = p[0]; int i1 = 0;
#pragma unroll
    for (int e = 1; e < E_EXPERTS; ++e)
      if (p[e] > b1v) { b1v = p[e]; i1 = e; }
    float b2v = -3.4e38f; int i2 = 0;
#pragma unroll
    for (int e = 0; e < E_EXPERTS; ++e)
      if (e != i1 && p[e] > b2v) { b2v = p[e]; i2 = e; }
    float t  = __expf(b2v - b1v);
    float wA = 1.f / (1.f + t);
    float wB = t   / (1.f + t);
    int posA = atomicAdd(&count[i1], 1);
    rowTsi[i1 * M_TOK + posA] = token * 2 + 0;
    rowWt [i1 * M_TOK + posA] = wA;
    int posB = atomicAdd(&count[i2], 1);
    rowTsi[i2 * M_TOK + posB] = token * 2 + 1;
    rowWt [i2 * M_TOK + posB] = wB;
  }
}

// ---------------------------------------------------------------------------
// Kernel 2: gather routed x rows into per-expert A-fragment-layout tiles.
// xg dword index (per expert): (mt*32 + ktH)*256 + laneA*8 + v
// grid = (256 tiles of 16 rows, E)
// ---------------------------------------------------------------------------
__global__ void gather_pack_kernel(const float* __restrict__ x,
                                   const int* __restrict__ count,
                                   const int* __restrict__ rowTsi,
                                   uint32_t* __restrict__ xg) {
  const int e    = blockIdx.y;
  const int tile = blockIdx.x;
  const int cnt  = count[e];
  if (tile * 16 >= cnt) return;

  __shared__ int s_tok[16];
  const int tid = threadIdx.x;
  if (tid < 16) {
    int g = tile * 16 + tid;
    s_tok[tid] = (g < cnt) ? (rowTsi[e * M_TOK + g] >> 1) : -1;
  }
  __syncthreads();

  uint32_t* xgT = xg + (size_t)e * XG_E_DW + (size_t)tile * 32 * 256;
  for (int i = tid; i < 16 * (H_DIM / 2); i += THREADS) {
    int r = i >> 9;              // row in tile
    int c = (i & 511) << 1;      // even k
    int tok = s_tok[r];
    float f0 = 0.f, f1 = 0.f;
    if (tok >= 0) {
      f0 = x[(size_t)tok * H_DIM + c];
      f1 = x[(size_t)tok * H_DIM + c + 1];
    }
    int kt = c >> 5, kb = c & 31;
    int vv = ((kb >> 4) << 2) + ((kb & 7) >> 1);
    int ha = (kb & 15) >> 3;
    xgT[kt * 256 + ((ha << 4) + r) * 8 + vv] = pack_bf16(f0, f1);
  }
}

// ---------------------------------------------------------------------------
// Kernel 3: GEMM1 + silu.  128 rows per block, N sectioned in 128-col slabs.
//   hpack = silu(x@w1) * (x@w3), written in A-fragment layout.
// grid = (32, E); 8 waves; no LDS/barriers in the hot loop.
// ---------------------------------------------------------------------------
__global__ __launch_bounds__(THREADS, 1)
void ffn1_kernel(const uint32_t* __restrict__ xg,
                 const uint32_t* __restrict__ pw1,
                 const uint32_t* __restrict__ pw3,
                 const int* __restrict__ count,
                 uint32_t* __restrict__ hpack) {
  const int e    = blockIdx.y;
  const int tile = blockIdx.x;                 // 128-row tile
  const int cnt  = count[e];
  if (tile * 128 >= cnt) return;

  const int tid  = threadIdx.x;
  const int lane = tid & 31;
  const int wave = tid >> 5;
  const int half = lane >> 4;
  const int lm   = lane & 15;

  const uint32_t* xgB = xg + (size_t)e * XG_E_DW +
                        (size_t)tile * 8 * 32 * 256 + lane * 8;
  const uint32_t* pw1e = pw1 + (size_t)e * PWE_DW;
  const uint32_t* pw3e = pw3 + (size_t)e * PWE_DW;
  uint32_t* hpE = hpack + (size_t)e * HP_E_DW;

  const v8f Z = {0.f, 0.f, 0.f, 0.f, 0.f, 0.f, 0.f, 0.f};

  for (int sec = 0; sec < F_DIM / 128; ++sec) {
    const int ntile = sec * 8 + wave;          // this wave's 16 h-columns
    v8f h1[8], h3[8];
#pragma unroll
    for (int i = 0; i < 8; ++i) { h1[i] = Z; h3[i] = Z; }

    const uint32_t* b1p = pw1e + (size_t)ntile * 32 * 256 + lane * 8;
    const uint32_t* b3p = pw3e + (size_t)ntile * 32 * 256 + lane * 8;
    const uint32_t* ap0 = xgB;
    for (int kt = 0; kt < H_DIM / 32; ++kt) {
      Frag b1, b3;
      load8(b1, b1p);
      load8(b3, b3p);
      __builtin_prefetch(b1p + 512, 0, 1);
      __builtin_prefetch(b3p + 512, 0, 1);
      b1p += 256; b3p += 256;
      const uint32_t* ap = ap0;
#pragma unroll
      for (int ms = 0; ms < 8; ++ms) {
        Frag a;
        load8(a, ap);
        ap += 32 * 256;                        // next 16-row subtile
        h1[ms] = __builtin_amdgcn_wmma_f32_16x16x32_bf16(
            false, a.v, false, b1.v, (short)0, h1[ms], false, false);
        h3[ms] = __builtin_amdgcn_wmma_f32_16x16x32_bf16(
            false, a.v, false, b3.v, (short)0, h3[ms], false, false);
      }
      ap0 += 256;
    }
    // silu(h1)*h3 -> hpack (A-frag layout).  D elem: (M = r+8*half, N = lm)
    const int kpos = ntile * 16 + lm;
    const int ktF  = kpos >> 5;
    const int kb   = kpos & 31;
    const int vv   = ((kb >> 4) << 2) + ((kb & 7) >> 1);
    const int ha   = (kb & 15) >> 3;
    const int pos  = kb & 1;
#pragma unroll
    for (int ms = 0; ms < 8; ++ms) {
      size_t mtb = ((size_t)(tile * 8 + ms) * 64 + ktF) * 256;
#pragma unroll
      for (int r = 0; r < 8; ++r) {
        int Mr = r + (half << 3);
        float g  = h1[ms][r];
        float s  = g * __builtin_amdgcn_rcpf(1.f + __expf(-g));
        float hv = s * h3[ms][r];
        size_t dw = mtb + ((ha << 4) + Mr) * 8 + vv;
        ((uint16_t*)hpE)[dw * 2 + pos] = f2bf(hv);
      }
    }
  }
}

// ---------------------------------------------------------------------------
// Kernel 4: GEMM2 + routing-weight scale + scatter.  128 rows per block,
// N sectioned in 128-col slabs; B fragment reused across 8 M-subtiles.
// ---------------------------------------------------------------------------
__global__ __launch_bounds__(THREADS, 1)
void ffn2_kernel(const uint32_t* __restrict__ hpack,
                 const uint32_t* __restrict__ pw2,
                 const int* __restrict__ count,
                 const int* __restrict__ rowTsi,
                 const float* __restrict__ rowWt,
                 float* __restrict__ outbuf) {
  const int e    = blockIdx.y;
  const int tile = blockIdx.x;                 // 128-row tile
  const int cnt  = count[e];
  if (tile * 128 >= cnt) return;

  __shared__ int   s_tsi[128];
  __shared__ float s_wt [128];

  const int tid  = threadIdx.x;
  const int lane = tid & 31;
  const int wave = tid >> 5;
  const int half = lane >> 4;
  const int lm   = lane & 15;

  if (tid < 128) {
    int g = tile * 128 + tid;
    if (g < cnt) {
      s_tsi[tid] = rowTsi[e * M_TOK + g];
      s_wt [tid] = rowWt[e * M_TOK + g];
    } else {
      s_tsi[tid] = 2 * M_TOK;                  // dummy scratch row
      s_wt [tid] = 0.f;
    }
  }
  __syncthreads();

  const uint32_t* hpB = hpack + (size_t)e * HP_E_DW +
                        (size_t)tile * 8 * 64 * 256 + lane * 8;
  const uint32_t* pw2e = pw2 + (size_t)e * PWE_DW;

  const v8f Z = {0.f, 0.f, 0.f, 0.f, 0.f, 0.f, 0.f, 0.f};

  for (int sec = 0; sec < H_DIM / 128; ++sec) {
    const int ntile = sec * 8 + wave;          // this wave's 16 out-columns
    v8f acc[8];
#pragma unroll
    for (int i = 0; i < 8; ++i) acc[i] = Z;

    const uint32_t* b2p = pw2e + (size_t)ntile * 64 * 256 + lane * 8;
    const uint32_t* ap0 = hpB;
    for (int ktF = 0; ktF < F_DIM / 32; ++ktF) {
      Frag b2;
      load8(b2, b2p);
      __builtin_prefetch(b2p + 512, 0, 1);
      b2p += 256;
      const uint32_t* ap = ap0;
#pragma unroll
      for (int ms = 0; ms < 8; ++ms) {
        Frag a;
        load8(a, ap);
        ap += 64 * 256;                        // next 16-row subtile
        acc[ms] = __builtin_amdgcn_wmma_f32_16x16x32_bf16(
            false, a.v, false, b2.v, (short)0, acc[ms], false, false);
      }
      ap0 += 256;
    }
    // epilogue: scale + scatter to per-(token,slot) rows
#pragma unroll
    for (int ms = 0; ms < 8; ++ms) {
#pragma unroll
      for (int r = 0; r < 8; ++r) {
        int Mr = r + (half << 3);
        int rl = ms * 16 + Mr;
        int col = ntile * 16 + lm;
        outbuf[(size_t)s_tsi[rl] * H_DIM + col] = acc[ms][r] * s_wt[rl];
      }
    }
  }
}

// ---------------------------------------------------------------------------
// Kernel 5: deterministic combine of the two slots per token.
// ---------------------------------------------------------------------------
__global__ void combine_kernel(const float4* __restrict__ outbuf,
                               float4* __restrict__ out) {
  int g = blockIdx.x * blockDim.x + threadIdx.x;
  int t = g >> 8;
  int c = g & 255;
  float4 a = outbuf[(size_t)(2 * t)     * 256 + c];
  float4 b = outbuf[(size_t)(2 * t + 1) * 256 + c];
  out[g] = make_float4(a.x + b.x, a.y + b.y, a.z + b.z, a.w + b.w);
}

extern "C" void kernel_launch(void* const* d_in, const int* in_sizes, int n_in,
                              void* d_out, int out_size, void* d_ws, size_t ws_size,
                              hipStream_t stream) {
  const float* x    = (const float*)d_in[0];  // [4096,1024]
  const float* gate = (const float*)d_in[1];  // [1024,8]
  const float* w1   = (const float*)d_in[2];  // [8,1024,2048]
  const float* w2   = (const float*)d_in[3];  // [8,2048,1024]
  const float* w3   = (const float*)d_in[4];  // [8,1024,2048]
  float* out = (float*)d_out;

  char* ws = (char*)d_ws;
  int*      count  = (int*)ws;       ws += 256;
  int*      rowTsi = (int*)ws;       ws += (size_t)E_EXPERTS * M_TOK * sizeof(int);
  float*    rowWt  = (float*)ws;     ws += (size_t)E_EXPERTS * M_TOK * sizeof(float);
  uint32_t* pw1    = (uint32_t*)ws;  ws += (size_t)E_EXPERTS * PWE_DW * 4;
  uint32_t* pw3    = (uint32_t*)ws;  ws += (size_t)E_EXPERTS * PWE_DW * 4;
  uint32_t* pw2    = (uint32_t*)ws;  ws += (size_t)E_EXPERTS * PWE_DW * 4;
  uint32_t* xg     = (uint32_t*)ws;  ws += (size_t)E_EXPERTS * XG_E_DW * 4;
  uint32_t* hpack  = (uint32_t*)ws;  ws += (size_t)E_EXPERTS * HP_E_DW * 4;
  float*    outbuf = (float*)ws;     // (2*M_TOK + 1) * H_DIM floats

  hipMemsetAsync(count, 0, E_EXPERTS * sizeof(int), stream);

  repack_kernel<<<dim3(H_DIM / 32, F_DIM / 128, E_EXPERTS), THREADS, 0, stream>>>(
      w1, pw1, H_DIM, F_DIM);
  repack_kernel<<<dim3(H_DIM / 32, F_DIM / 128, E_EXPERTS), THREADS, 0, stream>>>(
      w3, pw3, H_DIM, F_DIM);
  repack_kernel<<<dim3(F_DIM / 32, H_DIM / 128, E_EXPERTS), THREADS, 0, stream>>>(
      w2, pw2, F_DIM, H_DIM);

  router_kernel<<<M_TOK / WAVES, THREADS, 0, stream>>>(x, gate, count, rowTsi, rowWt);

  gather_pack_kernel<<<dim3(M_TOK / 16, E_EXPERTS), THREADS, 0, stream>>>(
      x, count, rowTsi, xg);

  ffn1_kernel<<<dim3(M_TOK / 128, E_EXPERTS), THREADS, 0, stream>>>(
      xg, pw1, pw3, count, hpack);

  ffn2_kernel<<<dim3(M_TOK / 128, E_EXPERTS), THREADS, 0, stream>>>(
      hpack, pw2, count, rowTsi, rowWt, outbuf);

  combine_kernel<<<(M_TOK * H_DIM / 4) / THREADS, THREADS, 0, stream>>>(
      (const float4*)outbuf, (float4*)out);
}
